// HGT_40939628265927
// MI455X (gfx1250) — matrix operand order repaired and checked by reading, not compiled, not run on hardware
//
#include <hip/hip_runtime.h>

// ---------------------------------------------------------------------------
// HGT for MI455X (gfx1250): bf16 WMMA GEMMs + folded relation transforms +
// atomic segment-softmax edge aggregation.
// ---------------------------------------------------------------------------

typedef __bf16 bf16x16 __attribute__((ext_vector_type(16)));
typedef float  f32x8   __attribute__((ext_vector_type(8)));

union WFrag { bf16x16 v; unsigned u[8]; };

enum { ACT_NONE = 0, ACT_RELU = 1, ACT_SIGMOID = 2, ACT_SKIPMIX = 3 };

__device__ __forceinline__ unsigned short f2bf(float f) {
    unsigned u = __float_as_uint(f);
    u += 0x7FFFu + ((u >> 16) & 1u);       // round-to-nearest-even
    return (unsigned short)(u >> 16);
}

__device__ __forceinline__ float gelu_f(float x) {
    const float k0 = 0.7978845608028654f;  // sqrt(2/pi)
    const float k1 = 0.044715f;
    return 0.5f * x * (1.f + tanhf(k0 * (x + k1 * x * x * x)));
}

__device__ __forceinline__ void atomicMaxF(float* addr, float val) {
    unsigned* ua = (unsigned*)addr;
    unsigned old = __float_as_uint(*addr);
    while (__uint_as_float(old) < val) {
        unsigned assumed = old;
        old = atomicCAS(ua, assumed, __float_as_uint(val));
        if (old == assumed) break;
    }
}

// ---------------------------------------------------------------------------
// Tiled WMMA GEMM: C[M,N] = epilogue( act(A)[M,128] @ B[128,N] + bias )
// Block = 256 threads (8 wave32), tile = 128 rows x 128 cols, K = 128.
// A and B staged in LDS as bf16; B stored K-contiguous (transposed) so every
// lane assembles its fragment with 4-byte ds reads.
// ---------------------------------------------------------------------------
__global__ void __launch_bounds__(256)
wmma_gemm128(const float* __restrict__ A, const float* __restrict__ B,
             const float* __restrict__ bias, float* __restrict__ C,
             int M, int N, int in_gelu, int out_act,
             const float* __restrict__ xold, const float* __restrict__ skipp)
{
    __shared__ unsigned short lA[128 * 128];   // [m][k]
    __shared__ unsigned short lB[128 * 128];   // [n][k]  (transposed)

    const int tid = threadIdx.x;
    const int m0  = blockIdx.x * 128;

    // --- stage A (optionally GELU'd) as bf16 ---
    for (int idx = tid; idx < 128 * 128; idx += 256) {
        int m = idx >> 7, k = idx & 127;
        float v = 0.f;
        if (m0 + m < M) v = A[(size_t)(m0 + m) * 128 + k];
        if (in_gelu) v = gelu_f(v);
        lA[idx] = f2bf(v);
    }
    // --- stage B transposed as bf16 (coalesced global reads over n) ---
    for (int idx = tid; idx < 128 * 128; idx += 256) {
        int n = idx & 127, k = idx >> 7;
        float v = (n < N) ? B[(size_t)k * N + n] : 0.f;
        lB[n * 128 + k] = f2bf(v);
    }
    __syncthreads();

    const int wave = tid >> 5;
    const int lane = tid & 31;
    const int mrow = (wave << 4) + (lane & 15);
    const int koffA = (lane < 16) ? 0 : 8;     // 16-bit A 16x32 K interleave
    const int koffB = (lane < 16) ? 0 : 8;     // B 32x16: same K interleave as A

    f32x8 zero = {0.f, 0.f, 0.f, 0.f, 0.f, 0.f, 0.f, 0.f};
    f32x8 acc[8];
#pragma unroll
    for (int n = 0; n < 8; ++n) acc[n] = zero;

#pragma unroll
    for (int kb = 0; kb < 128; kb += 32) {
        WFrag a;
#pragma unroll
        for (int j = 0; j < 4; ++j) {
            a.u[j]     = *(const unsigned*)&lA[mrow * 128 + kb + koffA + 2 * j];
            a.u[4 + j] = *(const unsigned*)&lA[mrow * 128 + kb + 16 + koffA + 2 * j];
        }
#pragma unroll
        for (int n = 0; n < 8; ++n) {
            WFrag b;
            const int ncol = (n << 4) + (lane & 15);
#pragma unroll
            for (int j = 0; j < 4; ++j) {
                b.u[j]     = *(const unsigned*)&lB[ncol * 128 + kb + koffB + 2 * j];
                b.u[4 + j] = *(const unsigned*)&lB[ncol * 128 + kb + 16 + koffB + 2 * j];
            }
            acc[n] = __builtin_amdgcn_wmma_f32_16x16x32_bf16(
                false, a.v, false, b.v, (short)0, acc[n], false, false);
        }
    }

    float alpha = 0.f;
    if (out_act == ACT_SKIPMIX) alpha = 1.f / (1.f + __expf(-(*skipp)));

#pragma unroll
    for (int n = 0; n < 8; ++n) {
        const int col = (n << 4) + (lane & 15);
#pragma unroll
        for (int i = 0; i < 8; ++i) {
            const int row = m0 + (wave << 4) + i + ((lane < 16) ? 0 : 8);
            if (row < M && col < N) {
                float v = acc[n][i] + bias[col];
                if (out_act == ACT_RELU)         v = fmaxf(v, 0.f);
                else if (out_act == ACT_SIGMOID) v = 1.f / (1.f + __expf(-v));
                else if (out_act == ACT_SKIPMIX)
                    v = alpha * v + (1.f - alpha) * xold[(size_t)row * 128 + col];
                C[(size_t)row * N + col] = v;
            }
        }
    }
}

// ---------------------------------------------------------------------------
// Fold block-diagonal per-head relation transform into projection weights:
//   Wout[c, h*16+e] = sum_d W[c, h*16+d] * R[h, d, e]
//   bout[h*16+e]    = sum_d b[h*16+d]    * R[h, d, e]
// ---------------------------------------------------------------------------
__global__ void __launch_bounds__(256)
fold_rel_kernel(const float* __restrict__ W, const float* __restrict__ b,
                const float* __restrict__ R, float* __restrict__ Wout,
                float* __restrict__ bout)
{
    int idx = blockIdx.x * 256 + threadIdx.x;
    if (idx < 128 * 128) {
        int c = idx >> 7, col = idx & 127, h = col >> 4, e = col & 15;
        float s = 0.f;
#pragma unroll
        for (int d = 0; d < 16; ++d)
            s += W[c * 128 + h * 16 + d] * R[h * 256 + d * 16 + e];
        Wout[idx] = s;
    }
    if (idx < 128) {
        int h = idx >> 4, e = idx & 15;
        float s = 0.f;
#pragma unroll
        for (int d = 0; d < 16; ++d)
            s += b[h * 16 + d] * R[h * 256 + d * 16 + e];
        bout[idx] = s;
    }
}

__global__ void __launch_bounds__(256)
fill_kernel(float* __restrict__ p, float v, int n)
{
    int i = blockIdx.x * 256 + threadIdx.x;
    if (i < n) p[i] = v;
}

__global__ void __launch_bounds__(256)
gather_relu_kernel(const float* __restrict__ emb, const int* __restrict__ idx,
                   float* __restrict__ out, int n)
{
    int i = blockIdx.x * 256 + threadIdx.x;
    if (i < n * 128) {
        int r = i >> 7, c = i & 127;
        out[i] = fmaxf(emb[(size_t)idx[r] * 128 + c], 0.f);
    }
}

// --- edge pass 1: logits + segment max ------------------------------------
__global__ void __launch_bounds__(256)
edge_logits_kernel(const float* __restrict__ q, const float* __restrict__ kr,
                   const int* __restrict__ src, const int* __restrict__ dst,
                   const float* __restrict__ prel, float* __restrict__ logits,
                   float* __restrict__ mx, int E)
{
    int i = blockIdx.x * 256 + threadIdx.x;
    if (i >= E * 8) return;
    int e = i >> 3, h = i & 7;
    int s = src[e], d = dst[e];
    const float4* qp = (const float4*)(q  + (size_t)d * 128 + h * 16);
    const float4* kp = (const float4*)(kr + (size_t)s * 128 + h * 16);
    float acc = 0.f;
#pragma unroll
    for (int j = 0; j < 4; ++j) {
        float4 a = qp[j], b = kp[j];
        acc += a.x * b.x + a.y * b.y + a.z * b.z + a.w * b.w;
    }
    float lgv = acc * prel[h] * 0.25f;   // scale = 1/sqrt(D) = 0.25
    logits[i] = lgv;
    atomicMaxF(&mx[(size_t)d * 8 + h], lgv);
}

// --- edge pass 2: exp + segment sum ---------------------------------------
__global__ void __launch_bounds__(256)
edge_expsum_kernel(float* __restrict__ logits, const float* __restrict__ mx,
                   const int* __restrict__ dst, float* __restrict__ sm, int E)
{
    int i = blockIdx.x * 256 + threadIdx.x;
    if (i >= E * 8) return;
    int e = i >> 3, h = i & 7;
    int d = dst[e];
    float w = __expf(logits[i] - mx[(size_t)d * 8 + h]);
    logits[i] = w;
    atomicAdd(&sm[(size_t)d * 8 + h], w);
}

// --- edge pass 3: normalized weighted scatter of vr -----------------------
__global__ void __launch_bounds__(256)
edge_scatter_kernel(const float* __restrict__ logits, const float* __restrict__ sm,
                    const float* __restrict__ vr, const int* __restrict__ src,
                    const int* __restrict__ dst, float* __restrict__ agg, int E)
{
    int i = blockIdx.x * 256 + threadIdx.x;
    if (i >= E * 8) return;
    int e = i >> 3, h = i & 7;
    int s = src[e], d = dst[e];
    float coef = logits[i] / sm[(size_t)d * 8 + h];
    const float4* vp = (const float4*)(vr + (size_t)s * 128 + h * 16);
    float* ap = agg + (size_t)d * 128 + h * 16;
#pragma unroll
    for (int j = 0; j < 4; ++j) {
        float4 v = vp[j];
        atomicAdd(&ap[4 * j + 0], v.x * coef);
        atomicAdd(&ap[4 * j + 1], v.y * coef);
        atomicAdd(&ap[4 * j + 2], v.z * coef);
        atomicAdd(&ap[4 * j + 3], v.w * coef);
    }
}

// ---------------------------------------------------------------------------
extern "C" void kernel_launch(void* const* d_in, const int* in_sizes, int n_in,
                              void* d_out, int out_size, void* d_ws, size_t ws_size,
                              hipStream_t stream)
{
    const int C = 128, H = 8;
    const int NPAT = in_sizes[0] / C;
    const int NICD = in_sizes[19];
    const int NNDC = in_sizes[20];
    const int E    = in_sizes[21];
    const int OUT  = in_sizes[18];
    const int sizes[3] = { NPAT, NICD, NNDC };

    const float* x_patient = (const float*)d_in[0];
    const float* w_in      = (const float*)d_in[1];
    const float* b_in      = (const float*)d_in[2];
    const float* emb_icd   = (const float*)d_in[3];
    const float* emb_ndc   = (const float*)d_in[4];
    const float* kw   = (const float*)d_in[5];
    const float* kb   = (const float*)d_in[6];
    const float* qw   = (const float*)d_in[7];
    const float* qbm  = (const float*)d_in[8];
    const float* vw   = (const float*)d_in[9];
    const float* vb   = (const float*)d_in[10];
    const float* aw   = (const float*)d_in[11];
    const float* ab   = (const float*)d_in[12];
    const float* skip = (const float*)d_in[13];
    const float* a_rel = (const float*)d_in[14];
    const float* m_rel = (const float*)d_in[15];
    const float* p_rel = (const float*)d_in[16];
    const float* w_out = (const float*)d_in[17];
    const float* b_out = (const float*)d_in[18];
    const int* x_icd = (const int*)d_in[19];
    const int* x_ndc = (const int*)d_in[20];
    const int* srcp[4] = { (const int*)d_in[21], (const int*)d_in[23],
                           (const int*)d_in[25], (const int*)d_in[27] };
    const int* dstp[4] = { (const int*)d_in[22], (const int*)d_in[24],
                           (const int*)d_in[26], (const int*)d_in[28] };
    const int ST[4] = {0, 1, 0, 2}, DT[4] = {1, 0, 2, 0};

    // ---- workspace carve-out (all 16B aligned) ----
    float* ws = (float*)d_ws;
    size_t off = 0;
    auto alloc = [&](size_t n) { float* p = ws + off; off += (n + 3) & ~(size_t)3; return p; };
    const size_t NX = (size_t)(NPAT + NICD + NNDC) * C;
    float* xa  = alloc(NX);
    float* xb  = alloc(NX);
    float* qqb = alloc(NX);
    float* agb = alloc(NX);
    float* krb = alloc((size_t)NPAT * C);
    float* vrb = alloc((size_t)NPAT * C);
    float* lg  = alloc((size_t)E * H);
    float* mx  = alloc((size_t)NPAT * H);
    float* sm  = alloc((size_t)NPAT * H);
    float* wkf = alloc(128 * 128); float* bkf = alloc(128);
    float* wvf = alloc(128 * 128); float* bvf = alloc(128);

    float* xc[3] = { xa, xa + (size_t)NPAT * C, xa + (size_t)(NPAT + NICD) * C };
    float* xn[3] = { xb, xb + (size_t)NPAT * C, xb + (size_t)(NPAT + NICD) * C };
    float* qq[3] = { qqb, qqb + (size_t)NPAT * C, qqb + (size_t)(NPAT + NICD) * C };
    float* ag[3] = { agb, agb + (size_t)NPAT * C, agb + (size_t)(NPAT + NICD) * C };

    auto gemm = [&](const float* A, const float* B, const float* bias, float* Cp,
                    int M, int N, int ingelu, int oact,
                    const float* xold, const float* sk) {
        wmma_gemm128<<<dim3((M + 127) / 128), 256, 0, stream>>>(
            A, B, bias, Cp, M, N, ingelu, oact, xold, sk);
    };
    auto fill = [&](float* p, float v, int n) {
        fill_kernel<<<dim3((n + 255) / 256), 256, 0, stream>>>(p, v, n);
    };

    // ---- input projection / embeddings ----
    gemm(x_patient, w_in, b_in, xc[0], NPAT, C, 0, ACT_RELU, nullptr, nullptr);
    gather_relu_kernel<<<dim3((NICD * C + 255) / 256), 256, 0, stream>>>(emb_icd, x_icd, xc[1], NICD);
    gather_relu_kernel<<<dim3((NNDC * C + 255) / 256), 256, 0, stream>>>(emb_ndc, x_ndc, xc[2], NNDC);

    const int egrid = (E * H + 255) / 256;

    for (int l = 0; l < 2; ++l) {
        // q projections for all node types
        for (int t = 0; t < 3; ++t)
            gemm(xc[t], qw + (size_t)(l * 3 + t) * C * C, qbm + (size_t)(l * 3 + t) * C,
                 qq[t], sizes[t], C, 0, ACT_NONE, nullptr, nullptr);
        fill(agb, 0.f, (int)NX);

        for (int r = 0; r < 4; ++r) {
            const int st = ST[r], dt = DT[r];
            const float* Rk = a_rel + (size_t)(l * 4 + r) * H * 16 * 16;
            const float* Rv = m_rel + (size_t)(l * 4 + r) * H * 16 * 16;
            // fold a_rel/m_rel into kw/vw (block-diagonal per head)
            fold_rel_kernel<<<64, 256, 0, stream>>>(
                kw + (size_t)(l * 3 + st) * C * C, kb + (size_t)(l * 3 + st) * C, Rk, wkf, bkf);
            fold_rel_kernel<<<64, 256, 0, stream>>>(
                vw + (size_t)(l * 3 + st) * C * C, vb + (size_t)(l * 3 + st) * C, Rv, wvf, bvf);
            gemm(xc[st], wkf, bkf, krb, sizes[st], C, 0, ACT_NONE, nullptr, nullptr);
            gemm(xc[st], wvf, bvf, vrb, sizes[st], C, 0, ACT_NONE, nullptr, nullptr);

            fill(mx, -1e30f, sizes[dt] * H);
            fill(sm, 0.f, sizes[dt] * H);
            edge_logits_kernel<<<egrid, 256, 0, stream>>>(
                qq[dt], krb, srcp[r], dstp[r], p_rel + (size_t)(l * 4 + r) * H, lg, mx, E);
            edge_expsum_kernel<<<egrid, 256, 0, stream>>>(lg, mx, dstp[r], sm, E);
            edge_scatter_kernel<<<egrid, 256, 0, stream>>>(lg, sm, vrb, srcp[r], dstp[r], ag[dt], E);
        }

        // out proj with GELU-on-input + learned skip mix
        for (int t = 0; t < 3; ++t)
            gemm(ag[t], aw + (size_t)(l * 3 + t) * C * C, ab + (size_t)(l * 3 + t) * C,
                 xn[t], sizes[t], C, 1, ACT_SKIPMIX, xc[t], skip + (l * 3 + t));
        for (int t = 0; t < 3; ++t) { float* tmp = xc[t]; xc[t] = xn[t]; xn[t] = tmp; }
    }

    // ---- final head ----
    gemm(xc[0], w_out, b_out, (float*)d_out, NPAT, OUT, 0, ACT_SIGMOID, nullptr, nullptr);
}
